// ConnectionTransformer_15530601742641
// MI455X (gfx1250) — compile-verified, matrix operand order checked by reading, not run on hardware
//
#include <hip/hip_runtime.h>
#include <stdint.h>

// ---------------------------------------------------------------------------
// ConnectionTransformer forward for MI455X (gfx1250, wave32, WMMA).
//
// Roofline strategy:
//  * Dominant cost: streaming W_source/W_target in the 6-step bilinear loop.
//    One-time fp32->bf16 transposed conversion (1.6 GB pass) halves per-step
//    traffic to 536 MB/step; each pair block is read from HBM exactly once
//    per step. bf16 intensity = 4 flop/byte -> WMMA trivially keeps up.
//  * All GEMMs: v_wmma_f32_16x16x32_bf16, fp32 accumulate. Waves within a
//    workgroup share the B (weight) column block -> WGP$ reuse (critical for
//    the 16 MB w_out operand in the 2048x32000x256 vocab GEMM).
//  * Bilinear: grid (j, i-quarter), 8 waves/WG; stage1 inter=Hs*Ws via WMMA
//    -> bf16 LDS -> stage2 influence += inter*Wt via WMMA (K=RANK=32).
// ---------------------------------------------------------------------------

typedef unsigned short ushort_t;
typedef __attribute__((ext_vector_type(16))) __bf16 v16bf;
typedef __attribute__((ext_vector_type(8)))  float  v8f;

#define D_MODEL 256
#define NSLOTS  128
#define RANK    32
#define NSTEPS  6
#define SEQ     512
#define BATCH   4
#define NVOCAB  32000
#define ISPLIT  4        // i-range split for the bilinear kernel

// ---------------- small device helpers ----------------

__device__ __forceinline__ ushort_t f2bf(float f) {
  union { float f; uint32_t u; } x; x.f = f;
  uint32_t r = x.u + 0x7FFFu + ((x.u >> 16) & 1u);   // round-to-nearest-even
  return (ushort_t)(r >> 16);
}

// Load one WMMA 16-bit operand tile (A: rows=M, or B: rows=N) from a
// row-major [16][ld] bf16 tile. p must already include row*ld + 8*kh.
// Elements 0..7 = cols [8kh,8kh+8), 8..15 = cols [16+8kh,24+8kh)  (ISA layout).
__device__ __forceinline__ v16bf ld_tile(const ushort_t* p) {
  union { uint4 q[2]; v16bf v; } u;
  u.q[0] = *(const uint4*)p;
  u.q[1] = *(const uint4*)(p + 16);
  return u.v;
}
__device__ __forceinline__ v16bf zero_tile() {
  union { uint4 q[2]; v16bf v; } u;
  u.q[0] = make_uint4(0u, 0u, 0u, 0u);
  u.q[1] = make_uint4(0u, 0u, 0u, 0u);
  return u.v;
}
__device__ __forceinline__ v8f zero8() {
  v8f c;
#pragma unroll
  for (int e = 0; e < 8; ++e) c[e] = 0.f;
  return c;
}
__device__ __forceinline__ v8f wmma_bf(v16bf a, v16bf b, v8f c) {
  return __builtin_amdgcn_wmma_f32_16x16x32_bf16(
      false, a, false, b, (short)0, c, false, false);
}
__device__ __forceinline__ float wave_sum(float x) {
#pragma unroll
  for (int m = 16; m > 0; m >>= 1) x += __shfl_xor(x, m, 32);
  return x;
}
__device__ __forceinline__ float wave_max(float x) {
#pragma unroll
  for (int m = 16; m > 0; m >>= 1) x = fmaxf(x, __shfl_xor(x, m, 32));
  return x;
}

// ---------------- elementwise / conversion kernels ----------------

__global__ void embed_kernel(const int* __restrict__ ids,
                             const float* __restrict__ tok,
                             const float* __restrict__ pos,
                             ushort_t* __restrict__ xbf) {
  int idx = blockIdx.x * 256 + threadIdx.x;       // over B*SEQ*D
  if (idx >= BATCH * SEQ * D_MODEL) return;
  int d  = idx & (D_MODEL - 1);
  int bs = idx >> 8;                              // D_MODEL == 256
  int s  = bs & (SEQ - 1);
  int id = ids[bs];
  xbf[idx] = f2bf(tok[(size_t)id * D_MODEL + d] + pos[s * D_MODEL + d]);
}

__global__ void cast_bf_kernel(const float* __restrict__ in,
                               ushort_t* __restrict__ out, size_t n) {
  size_t i = (size_t)blockIdx.x * 256 + threadIdx.x;
  if (i < n) out[i] = f2bf(in[i]);
}

// Per-pair transpose+cast: in fp32 [R][C] -> out bf16 [C][R]; one block/pair.
__global__ void transpose_cast_pair_kernel(const float* __restrict__ in,
                                           ushort_t* __restrict__ out,
                                           int R, int C) {
  extern __shared__ float tile[];                 // R*(C+1) floats (bank-padded)
  size_t pair = blockIdx.x;
  const float* pin = in + pair * (size_t)(R * C);
  ushort_t* pout   = out + pair * (size_t)(R * C);
  for (int e = threadIdx.x; e < R * C; e += blockDim.x) {
    int r = e / C, c = e - r * C;
    tile[r * (C + 1) + c] = pin[e];
  }
  __syncthreads();
  for (int e = threadIdx.x; e < R * C; e += blockDim.x) {
    int c = e / R, r = e - c * R;
    pout[e] = f2bf(tile[r * (C + 1) + c]);
  }
}

// bf16 [R][C] -> [C][R], batched over blockIdx.y
__global__ void transpose_bf_kernel(const ushort_t* __restrict__ in,
                                    ushort_t* __restrict__ out, int R, int C) {
  const ushort_t* pin = in + (size_t)blockIdx.y * R * C;
  ushort_t* pout      = out + (size_t)blockIdx.y * R * C;
  int e = blockIdx.x * 256 + threadIdx.x;
  if (e < R * C) {
    int r = e / C, c = e - r * C;
    pout[(size_t)c * R + r] = pin[e];
  }
}

// ---------------- generic batched WMMA GEMM: C[m,n] = sum_k A[m,k]*B[n,k] ----
// One wave computes a 16x64 C block (4 N-tiles). M%16==0, N%64==0, K%32==0.
// Decode: the 8 waves of a block take 8 consecutive M-tiles with the SAME
// N column block -> the (large) B operand block hits the WGP$ 7/8 times.

__global__ __launch_bounds__(256) void gemm_bt_kernel(
    const ushort_t* __restrict__ A, int lda, size_t strideA,
    const ushort_t* __restrict__ B, int ldb, size_t strideB,
    float* __restrict__ Cf, ushort_t* __restrict__ Cb, int ldc, size_t strideC,
    int tilesM, int tilesN4, int kChunks) {
  int wave = threadIdx.x >> 5;
  int lane = threadIdx.x & 31;
  int row  = lane & 15;
  int kh   = lane >> 4;
  long tile  = (long)blockIdx.x * 8 + wave;
  long total = (long)tilesM * tilesN4;
  if (tile >= total) return;                       // wave-uniform
  int tn = (int)(tile / tilesM);                   // waves in a block share tn
  int tm = (int)(tile % tilesM);
  const ushort_t* Ab = A + (size_t)blockIdx.z * strideA + (size_t)(tm * 16) * lda;
  const ushort_t* Bb = B + (size_t)blockIdx.z * strideB + (size_t)(tn * 64) * ldb;
  v8f c0 = zero8(), c1 = zero8(), c2 = zero8(), c3 = zero8();
  for (int kc = 0; kc < kChunks; ++kc) {
    const ushort_t* ap = Ab + (size_t)row * lda + kc * 32 + 8 * kh;
    v16bf a = ld_tile(ap);
    const ushort_t* bp = Bb + (size_t)row * ldb + kc * 32 + 8 * kh;
    v16bf b0 = ld_tile(bp);
    v16bf b1 = ld_tile(bp + 16 * (size_t)ldb);
    v16bf b2 = ld_tile(bp + 32 * (size_t)ldb);
    v16bf b3 = ld_tile(bp + 48 * (size_t)ldb);
    c0 = wmma_bf(a, b0, c0);
    c1 = wmma_bf(a, b1, c1);
    c2 = wmma_bf(a, b2, c2);
    c3 = wmma_bf(a, b3, c3);
  }
  size_t base = (size_t)blockIdx.z * strideC + (size_t)(tm * 16) * ldc + tn * 64;
#pragma unroll
  for (int v = 0; v < 8; ++v) {
    int m = v + 8 * kh;
    size_t o = base + (size_t)m * ldc + row;
    if (Cf) {
      Cf[o] = c0[v]; Cf[o + 16] = c1[v]; Cf[o + 32] = c2[v]; Cf[o + 48] = c3[v];
    } else {
      Cb[o] = f2bf(c0[v]); Cb[o + 16] = f2bf(c1[v]);
      Cb[o + 32] = f2bf(c2[v]); Cb[o + 48] = f2bf(c3[v]);
    }
  }
}

// ---------------- attention helper kernels ----------------

// Row softmax over NSLOTS cols. Optional mask (A *= m[row]), optional bf16 out.
__global__ void softmax_kernel(float* __restrict__ S, const int* __restrict__ mask,
                               ushort_t* __restrict__ outBf, int rows, float scale) {
  int wave = threadIdx.x >> 5, lane = threadIdx.x & 31;
  int row = blockIdx.x * 8 + wave;
  if (row >= rows) return;
  float v[4];
  float mx = -3.0e38f;
#pragma unroll
  for (int q = 0; q < 4; ++q) {
    v[q] = S[(size_t)row * NSLOTS + lane + 32 * q] * scale;
    mx = fmaxf(mx, v[q]);
  }
  mx = wave_max(mx);
  float sum = 0.f;
#pragma unroll
  for (int q = 0; q < 4; ++q) { v[q] = __expf(v[q] - mx); sum += v[q]; }
  sum = wave_sum(sum);
  float inv = 1.f / sum;
  float m = mask ? (float)mask[row] : 1.f;
#pragma unroll
  for (int q = 0; q < 4; ++q) {
    float p = v[q] * inv * m;
    size_t o = (size_t)row * NSLOTS + lane + 32 * q;
    if (outBf) outBf[o] = f2bf(p);
    else       S[o] = p;
  }
}

// Column (over s) normalization of A[b][s][k]; writes AT_bf[b][k][s].
__global__ void colnorm_kernel(const float* __restrict__ A,
                               ushort_t* __restrict__ ATbf) {
  int wave = threadIdx.x >> 5, lane = threadIdx.x & 31;
  int id = blockIdx.x * 8 + wave;                  // b*NSLOTS + k
  if (id >= BATCH * NSLOTS) return;
  int b = id >> 7, k = id & (NSLOTS - 1);
  const float* Ab = A + (size_t)b * SEQ * NSLOTS;
  float sum = 0.f;
  for (int s = lane; s < SEQ; s += 32) sum += Ab[(size_t)s * NSLOTS + k];
  sum = wave_sum(sum);
  float inv = 1.f / (sum + 1e-8f);
  ushort_t* o = ATbf + ((size_t)b * NSLOTS + k) * SEQ;
  for (int s = lane; s < SEQ; s += 32)
    o[s] = f2bf(Ab[(size_t)s * NSLOTS + k] * inv);
}

__global__ void addH_kernel(const float* __restrict__ H,
                            const float* __restrict__ IR,
                            float* __restrict__ Hs, ushort_t* __restrict__ Hsbf) {
  int idx = blockIdx.x * 256 + threadIdx.x;
  if (idx >= BATCH * NSLOTS * D_MODEL) return;
  float v = H[idx & (NSLOTS * D_MODEL - 1)] + IR[idx];
  Hs[idx] = v;
  Hsbf[idx] = f2bf(v);
}

// Hs = LN(Hs + relu(sum of ISPLIT influence partials)); wave per (b,slot).
__global__ void update_ln_kernel(float* __restrict__ Hs, ushort_t* __restrict__ Hsbf,
                                 const float* __restrict__ infl,
                                 const float* __restrict__ gamma,
                                 const float* __restrict__ beta) {
  int wave = threadIdx.x >> 5, lane = threadIdx.x & 31;
  int id = blockIdx.x * 8 + wave;                  // over B*NSLOTS
  if (id >= BATCH * NSLOTS) return;
  size_t base = (size_t)id * D_MODEL;
  const size_t part = (size_t)BATCH * NSLOTS * D_MODEL;
  float u[8];
  float sum = 0.f, sq = 0.f;
#pragma unroll
  for (int e = 0; e < 8; ++e) {
    int d = lane * 8 + e;
    float inf = 0.f;
#pragma unroll
    for (int p = 0; p < ISPLIT; ++p) inf += infl[p * part + base + d];
    float val = Hs[base + d] + fmaxf(inf, 0.f);
    u[e] = val; sum += val; sq += val * val;
  }
  sum = wave_sum(sum);
  sq  = wave_sum(sq);
  float mean = sum * (1.f / D_MODEL);
  float var  = sq * (1.f / D_MODEL) - mean * mean;
  float rs = rsqrtf(var + 1e-5f);
#pragma unroll
  for (int e = 0; e < 8; ++e) {
    int d = lane * 8 + e;
    float o = (u[e] - mean) * rs * gamma[d] + beta[d];
    Hs[base + d] = o;
    Hsbf[base + d] = f2bf(o);
  }
}

// ---------------- bilinear all-pairs slot mixing (the hot loop) -------------
// grid = (j=0..127, h=0..ISPLIT-1); block = 8 waves; wave w owns d-slice
// [32w, 32w+32). Each (h) quarter covers 32 i's = 2 blocks of 16.
// Stage 1 (per wave, 2 i's per block): inter[b,r] = Hs[b,i,:]*WsT[i,j]^T via
//   8x2 WMMAs -> bf16 tile in LDS (i==j -> zero tile = pair mask).
// Stage 2 (per wave, all 16 i's): acc += inter_i * WtT[i,j,dslice]^T (K=32).

__global__ __launch_bounds__(256) void bilinear_kernel(
    const ushort_t* __restrict__ Hsbf,   // [B][NSLOTS][D]
    const ushort_t* __restrict__ WsT,    // [i][j][RANK][D]
    const ushort_t* __restrict__ WtT,    // [i][j][D][RANK]
    float* __restrict__ infl) {          // [ISPLIT][B][NSLOTS][D]
  __shared__ ushort_t sInter[16 * 16 * 32];      // 16 i-slots x [16][32] bf16
  int j = blockIdx.x;
  int h = blockIdx.y;
  int wave = threadIdx.x >> 5, lane = threadIdx.x & 31;
  int row = lane & 15, kh = lane >> 4;
  int d0 = wave * 32;
  v8f acc0 = zero8(), acc1 = zero8();

  for (int blk = 0; blk < (NSLOTS / ISPLIT) / 16; ++blk) {
    int ibase = h * (NSLOTS / ISPLIT) + blk * 16;
    // ---- stage 1: compute inter for 16 i's (2 per wave) ----
#pragma unroll
    for (int t = 0; t < 2; ++t) {
      int il = wave + t * 8;
      int i = ibase + il;
      v8f c0 = zero8(), c1 = zero8();
      if (i != j) {
        const ushort_t* ws = WsT + ((size_t)i * NSLOTS + j) * (RANK * D_MODEL);
        const ushort_t* ab = Hsbf + ((size_t)row * NSLOTS + i) * D_MODEL;
        __builtin_prefetch(WsT + ((size_t)(i + 8) * NSLOTS + j) * (RANK * D_MODEL), 0, 1);
#pragma unroll
        for (int kc = 0; kc < 8; ++kc) {
          v16bf a = (row < BATCH) ? ld_tile(ab + kc * 32 + 8 * kh) : zero_tile();
          v16bf b0 = ld_tile(ws + (size_t)row * D_MODEL + kc * 32 + 8 * kh);
          v16bf b1 = ld_tile(ws + (size_t)(16 + row) * D_MODEL + kc * 32 + 8 * kh);
          c0 = wmma_bf(a, b0, c0);
          c1 = wmma_bf(a, b1, c1);
        }
      }
      ushort_t* s = sInter + il * 512;             // [16][32] row-major
#pragma unroll
      for (int v = 0; v < 8; ++v) {
        int m = v + 8 * kh;
        s[m * 32 + row]      = f2bf(c0[v]);        // r cols 0..15
        s[m * 32 + 16 + row] = f2bf(c1[v]);        // r cols 16..31
      }
    }
    __syncthreads();
    // ---- stage 2: accumulate influence for this wave's d-slice ----
#pragma unroll 4
    for (int il = 0; il < 16; ++il) {
      int i = ibase + il;
      const ushort_t* sA = sInter + il * 512;
      v16bf a = ld_tile(sA + row * 32 + 8 * kh);   // K = RANK = 32, one chunk
      const ushort_t* wt = WtT + ((size_t)i * NSLOTS + j) * (D_MODEL * RANK);
      __builtin_prefetch(WtT + ((size_t)(i + 1) * NSLOTS + j) * (D_MODEL * RANK) + d0 * RANK, 0, 1);
      v16bf b0 = ld_tile(wt + (size_t)(d0 + row) * RANK + 8 * kh);
      v16bf b1 = ld_tile(wt + (size_t)(d0 + 16 + row) * RANK + 8 * kh);
      acc0 = wmma_bf(a, b0, acc0);
      acc1 = wmma_bf(a, b1, acc1);
    }
    __syncthreads();
  }
  // write rows b=0..3 (m = v + 8*kh, valid only kh==0, v<4)
  if (kh == 0) {
#pragma unroll
    for (int v = 0; v < BATCH; ++v) {
      size_t o = (((size_t)h * BATCH + v) * NSLOTS + j) * D_MODEL + d0;
      infl[o + row]      = acc0[v];
      infl[o + 16 + row] = acc1[v];
    }
  }
}

// ---------------- host launcher ----------------

static inline void launch_gemm(hipStream_t s,
                               const ushort_t* A, int lda, size_t sA,
                               const ushort_t* B, int ldb, size_t sB,
                               float* Cf, ushort_t* Cb, int ldc, size_t sC,
                               int M, int N, int K, int batch) {
  int tilesM = M / 16, tilesN4 = N / 64, kC = K / 32;
  long total = (long)tilesM * tilesN4;
  dim3 g((unsigned)((total + 7) / 8), 1, (unsigned)batch);
  gemm_bt_kernel<<<g, 256, 0, s>>>(A, lda, sA, B, ldb, sB, Cf, Cb, ldc, sC,
                                   tilesM, tilesN4, kC);
}

extern "C" void kernel_launch(void* const* d_in, const int* in_sizes, int n_in,
                              void* d_out, int out_size, void* d_ws, size_t ws_size,
                              hipStream_t stream) {
  (void)in_sizes; (void)n_in; (void)out_size; (void)ws_size;
  const int*   ids      = (const int*)d_in[0];
  const int*   amask    = (const int*)d_in[1];
  const float* tok      = (const float*)d_in[2];
  const float* pos      = (const float*)d_in[3];
  const float* H        = (const float*)d_in[4];
  const float* Wsrc     = (const float*)d_in[5];
  const float* Wtgt     = (const float*)d_in[6];
  const float* wq_in    = (const float*)d_in[7];
  const float* wk_slots = (const float*)d_in[8];
  const float* wv_in    = (const float*)d_in[9];
  const float* wq_out   = (const float*)d_in[10];
  const float* wk_fin   = (const float*)d_in[11];
  const float* wv_fin   = (const float*)d_in[12];
  const float* w_out    = (const float*)d_in[13];
  const float* ln_g     = (const float*)d_in[14];
  const float* ln_b     = (const float*)d_in[15];
  float* logits = (float*)d_out;

  const float scale = 0.0625f;   // 1/sqrt(256)
  char* wsp = (char*)d_ws;
  size_t off = 0;
  auto alloc = [&](size_t bytes) -> char* {
    char* p = wsp + off;
    off = (off + bytes + 255) & ~(size_t)255;
    return p;
  };
  const size_t PAIR_ELEMS = (size_t)NSLOTS * NSLOTS * RANK * D_MODEL;  // 134M
  ushort_t* WsT = (ushort_t*)alloc(PAIR_ELEMS * 2);
  ushort_t* WtT = (ushort_t*)alloc(PAIR_ELEMS * 2);
  ushort_t* xbf = (ushort_t*)alloc((size_t)BATCH * SEQ * D_MODEL * 2);
  ushort_t* wqin_bf    = (ushort_t*)alloc(D_MODEL * D_MODEL * 2);
  ushort_t* wkslots_bf = (ushort_t*)alloc(D_MODEL * D_MODEL * 2);
  ushort_t* wvin_bf    = (ushort_t*)alloc(D_MODEL * D_MODEL * 2);
  ushort_t* wqout_bf   = (ushort_t*)alloc(D_MODEL * D_MODEL * 2);
  ushort_t* wkfin_bf   = (ushort_t*)alloc(D_MODEL * D_MODEL * 2);
  ushort_t* wvfin_bf   = (ushort_t*)alloc(D_MODEL * D_MODEL * 2);
  ushort_t* Hbf    = (ushort_t*)alloc(NSLOTS * D_MODEL * 2);
  ushort_t* woutbf = (ushort_t*)alloc((size_t)NVOCAB * D_MODEL * 2);
  ushort_t* Qibf = (ushort_t*)alloc((size_t)BATCH * SEQ * D_MODEL * 2);
  ushort_t* Vibf = (ushort_t*)alloc((size_t)BATCH * SEQ * D_MODEL * 2);
  ushort_t* Qobf = (ushort_t*)alloc((size_t)BATCH * SEQ * D_MODEL * 2);
  ushort_t* Kslotsbf = (ushort_t*)alloc(NSLOTS * D_MODEL * 2);
  float*    S1   = (float*)alloc((size_t)BATCH * SEQ * NSLOTS * 4);
  ushort_t* ATbf = (ushort_t*)alloc((size_t)BATCH * NSLOTS * SEQ * 2);
  ushort_t* ViTbf = (ushort_t*)alloc((size_t)BATCH * D_MODEL * SEQ * 2);
  float*    IR   = (float*)alloc((size_t)BATCH * NSLOTS * D_MODEL * 4);
  float*    Hs   = (float*)alloc((size_t)BATCH * NSLOTS * D_MODEL * 4);
  ushort_t* Hsbf = (ushort_t*)alloc((size_t)BATCH * NSLOTS * D_MODEL * 2);
  float*    infl = (float*)alloc((size_t)ISPLIT * BATCH * NSLOTS * D_MODEL * 4);
  ushort_t* Kfbf  = (ushort_t*)alloc((size_t)BATCH * NSLOTS * D_MODEL * 2);
  ushort_t* Vfbf  = (ushort_t*)alloc((size_t)BATCH * NSLOTS * D_MODEL * 2);
  ushort_t* VfTbf = (ushort_t*)alloc((size_t)BATCH * D_MODEL * NSLOTS * 2);
  float*    S2    = (float*)alloc((size_t)BATCH * SEQ * NSLOTS * 4);
  ushort_t* Aebf  = (ushort_t*)alloc((size_t)BATCH * SEQ * NSLOTS * 2);
  ushort_t* Ybf   = (ushort_t*)alloc((size_t)BATCH * SEQ * D_MODEL * 2);

  auto castbf = [&](const float* src, ushort_t* dst, size_t n) {
    cast_bf_kernel<<<dim3((unsigned)((n + 255) / 256)), 256, 0, stream>>>(src, dst, n);
  };

  // --- one-time (per launch) weight conversions ---
  castbf(wq_in, wqin_bf, D_MODEL * D_MODEL);
  castbf(wk_slots, wkslots_bf, D_MODEL * D_MODEL);
  castbf(wv_in, wvin_bf, D_MODEL * D_MODEL);
  castbf(wq_out, wqout_bf, D_MODEL * D_MODEL);
  castbf(wk_fin, wkfin_bf, D_MODEL * D_MODEL);
  castbf(wv_fin, wvfin_bf, D_MODEL * D_MODEL);
  castbf(H, Hbf, NSLOTS * D_MODEL);
  castbf(w_out, woutbf, (size_t)NVOCAB * D_MODEL);
  transpose_cast_pair_kernel<<<NSLOTS * NSLOTS, 256, D_MODEL * (RANK + 1) * 4, stream>>>(
      Wsrc, WsT, D_MODEL, RANK);     // [d][r] -> [r][d]
  transpose_cast_pair_kernel<<<NSLOTS * NSLOTS, 256, RANK * (D_MODEL + 1) * 4, stream>>>(
      Wtgt, WtT, RANK, D_MODEL);     // [r][d] -> [d][r]

  // --- embeddings + input projections ---
  embed_kernel<<<(BATCH * SEQ * D_MODEL) / 256, 256, 0, stream>>>(ids, tok, pos, xbf);
  launch_gemm(stream, xbf, D_MODEL, 0, wqin_bf, D_MODEL, 0,
              nullptr, Qibf, D_MODEL, 0, BATCH * SEQ, D_MODEL, D_MODEL, 1);
  launch_gemm(stream, xbf, D_MODEL, 0, wvin_bf, D_MODEL, 0,
              nullptr, Vibf, D_MODEL, 0, BATCH * SEQ, D_MODEL, D_MODEL, 1);
  launch_gemm(stream, xbf, D_MODEL, 0, wqout_bf, D_MODEL, 0,
              nullptr, Qobf, D_MODEL, 0, BATCH * SEQ, D_MODEL, D_MODEL, 1);
  launch_gemm(stream, Hbf, D_MODEL, 0, wkslots_bf, D_MODEL, 0,
              nullptr, Kslotsbf, D_MODEL, 0, NSLOTS, D_MODEL, D_MODEL, 1);

  // --- compression cross-attention ---
  launch_gemm(stream, Qibf, D_MODEL, 0, Kslotsbf, D_MODEL, 0,
              S1, nullptr, NSLOTS, 0, BATCH * SEQ, NSLOTS, D_MODEL, 1);
  softmax_kernel<<<(BATCH * SEQ) / 8, 256, 0, stream>>>(S1, amask, nullptr,
                                                        BATCH * SEQ, scale);
  colnorm_kernel<<<(BATCH * NSLOTS) / 8, 256, 0, stream>>>(S1, ATbf);
  transpose_bf_kernel<<<dim3((SEQ * D_MODEL) / 256, BATCH), 256, 0, stream>>>(
      Vibf, ViTbf, SEQ, D_MODEL);
  launch_gemm(stream, ATbf, SEQ, (size_t)NSLOTS * SEQ,
              ViTbf, SEQ, (size_t)D_MODEL * SEQ,
              IR, nullptr, D_MODEL, (size_t)NSLOTS * D_MODEL,
              NSLOTS, D_MODEL, SEQ, BATCH);
  addH_kernel<<<(BATCH * NSLOTS * D_MODEL) / 256, 256, 0, stream>>>(H, IR, Hs, Hsbf);

  // --- 6 steps of bilinear slot mixing (HBM-bound hot loop) ---
  for (int step = 0; step < NSTEPS; ++step) {
    bilinear_kernel<<<dim3(NSLOTS, ISPLIT), 256, 0, stream>>>(Hsbf, WsT, WtT, infl);
    update_ln_kernel<<<(BATCH * NSLOTS) / 8, 256, 0, stream>>>(
        Hs, Hsbf, infl, ln_g + step * D_MODEL, ln_b + step * D_MODEL);
  }

  // --- expansion cross-attention + vocab projection ---
  launch_gemm(stream, Hsbf, D_MODEL, 0, wkfin_bf, D_MODEL, 0,
              nullptr, Kfbf, D_MODEL, 0, BATCH * NSLOTS, D_MODEL, D_MODEL, 1);
  launch_gemm(stream, Hsbf, D_MODEL, 0, wvfin_bf, D_MODEL, 0,
              nullptr, Vfbf, D_MODEL, 0, BATCH * NSLOTS, D_MODEL, D_MODEL, 1);
  transpose_bf_kernel<<<dim3((NSLOTS * D_MODEL) / 256, BATCH), 256, 0, stream>>>(
      Vfbf, VfTbf, NSLOTS, D_MODEL);
  launch_gemm(stream, Qobf, D_MODEL, (size_t)SEQ * D_MODEL,
              Kfbf, D_MODEL, (size_t)NSLOTS * D_MODEL,
              S2, nullptr, NSLOTS, (size_t)SEQ * NSLOTS,
              SEQ, NSLOTS, D_MODEL, BATCH);
  softmax_kernel<<<(BATCH * SEQ) / 8, 256, 0, stream>>>(S2, nullptr, Aebf,
                                                        BATCH * SEQ, scale);
  launch_gemm(stream, Aebf, NSLOTS, (size_t)SEQ * NSLOTS,
              VfTbf, NSLOTS, (size_t)D_MODEL * NSLOTS,
              nullptr, Ybf, D_MODEL, (size_t)SEQ * D_MODEL,
              SEQ, D_MODEL, NSLOTS, BATCH);
  launch_gemm(stream, Ybf, D_MODEL, 0, woutbf, D_MODEL, 0,
              logits, nullptr, NVOCAB, 0, BATCH * SEQ, NVOCAB, D_MODEL, 1);
}
